// KNNSoftmax_7533372637490
// MI455X (gfx1250) — compile-verified, matrix operand order checked by reading, not run on hardware
//
#include <hip/hip_runtime.h>
#include <hip/hip_bf16.h>

#define N_PTS   4096
#define D_DIM   1024
#define M_GRP   8
#define N_POS   7      /* M-1 */
#define N_NEG   4088   /* N-M */
#define ALPHA_C 30.0f

typedef __attribute__((ext_vector_type(16))) __bf16 v16bf;
typedef __attribute__((ext_vector_type(8)))  float  v8f;

// ---------------- small helpers ----------------

__global__ void zero_acc_kernel(float* acc) {
    if (threadIdx.x < 4) acc[threadIdx.x] = 0.0f;
}

// split fp32 -> bf16 hi + bf16 lo (residual), so 3 bf16 WMMAs reproduce ~fp32 dot
__global__ void cvt_hilo_kernel(const float* __restrict__ x,
                                __bf16* __restrict__ xhi, __bf16* __restrict__ xlo) {
    int idx = blockIdx.x * blockDim.x + threadIdx.x;
    int stride = gridDim.x * blockDim.x;
    for (int i = idx; i < N_PTS * D_DIM; i += stride) {
        float f = x[i];
        __bf16 h = (__bf16)f;
        xhi[i] = h;
        xlo[i] = (__bf16)(f - (float)h);
    }
}

__global__ void row_sq_kernel(const float* __restrict__ x, float* __restrict__ sq) {
    __shared__ float red[256];
    int row = blockIdx.x, tid = threadIdx.x;
    float s = 0.0f;
    for (int k = tid; k < D_DIM; k += 256) {
        float f = x[(size_t)row * D_DIM + k];
        s += f * f;
    }
    red[tid] = s; __syncthreads();
    for (int h = 128; h > 0; h >>= 1) {
        if (tid < h) red[tid] += red[tid + h];
        __syncthreads();
    }
    if (tid == 0) sq[row] = red[0];
}

// ---------------- WMMA Gram + distance + scatter ----------------
// 8 waves per block arranged 2 (M) x 4 (N); each wave computes a 32x32 output
// block = 2x2 WMMA tiles (4 v8f accumulators), reusing each A/B fragment twice.
// A fragment: lane L (0-15) holds row (mTile+L), K k0..k0+15; lanes 16-31 hold K k0+16..k0+31.
// B fragment (= X^T tile): identical pattern keyed by the column index.
__device__ __forceinline__ void scatter_tile(const v8f& c, int mBase, int n,
        const float* __restrict__ sq, const int* __restrict__ targets,
        float* __restrict__ posOut, float* __restrict__ negOut) {
    float sqn = sq[n];
    int   tn  = targets[n];
#pragma unroll
    for (int r = 0; r < 8; ++r) {
        int m = mBase + r;
        if (m == n) continue;
        float dist = sqrtf(fmaxf(sq[m] + sqn - 2.0f * c[r], 1e-12f));
        int tm = targets[m];
        if (tm == tn) {
            int slot = n - (tm << 3) - ((n > m) ? 1 : 0);   // positives in index order
            posOut[(size_t)m * N_POS + slot] = dist;
        } else {
            int slot = (tn > tm) ? (n - M_GRP) : n;         // negatives skip own block
            negOut[(size_t)m * N_NEG + slot] = dist;
        }
    }
}

#define WMMA_BF16(A, B, C) \
    __builtin_amdgcn_wmma_f32_16x16x32_bf16(false, (A), false, (B), (short)0, (C), false, false)

__global__ void __launch_bounds__(256) gram_dist_kernel(
        const __bf16* __restrict__ xhi, const __bf16* __restrict__ xlo,
        const float*  __restrict__ sq,  const int* __restrict__ targets,
        float* __restrict__ posOut, float* __restrict__ negOut) {
    int wave = threadIdx.x >> 5;
    int lane = threadIdx.x & 31;
    int wm = wave & 1;                 // 0..1  (M within block)
    int wn = wave >> 1;                // 0..3  (N within block)
    int bm = blockIdx.x & 63;          // 64 blocks along M (64 rows each)
    int bn = blockIdx.x >> 6;          // 32 blocks along N (128 cols each)

    int m0 = bm * 64 + wm * 32;        // wave's rows: [m0, m0+32)
    int n0 = bn * 128 + wn * 32;       // wave's cols: [n0, n0+32)

    int lr   = lane & 15;
    int kofs = (lane >> 4) * 16;

    const __bf16* pa0h = xhi + (size_t)(m0 + lr)      * D_DIM + kofs;
    const __bf16* pa1h = xhi + (size_t)(m0 + 16 + lr) * D_DIM + kofs;
    const __bf16* pa0l = xlo + (size_t)(m0 + lr)      * D_DIM + kofs;
    const __bf16* pa1l = xlo + (size_t)(m0 + 16 + lr) * D_DIM + kofs;
    const __bf16* pb0h = xhi + (size_t)(n0 + lr)      * D_DIM + kofs;
    const __bf16* pb1h = xhi + (size_t)(n0 + 16 + lr) * D_DIM + kofs;
    const __bf16* pb0l = xlo + (size_t)(n0 + lr)      * D_DIM + kofs;
    const __bf16* pb1l = xlo + (size_t)(n0 + 16 + lr) * D_DIM + kofs;

    v8f c00 = {}, c01 = {}, c10 = {}, c11 = {};
    for (int kt = 0; kt < D_DIM / 32; ++kt) {
        int k0 = kt * 32;
        v16bf a0h = *(const v16bf*)(pa0h + k0);
        v16bf a0l = *(const v16bf*)(pa0l + k0);
        v16bf a1h = *(const v16bf*)(pa1h + k0);
        v16bf a1l = *(const v16bf*)(pa1l + k0);
        v16bf b0h = *(const v16bf*)(pb0h + k0);
        v16bf b0l = *(const v16bf*)(pb0l + k0);
        v16bf b1h = *(const v16bf*)(pb1h + k0);
        v16bf b1l = *(const v16bf*)(pb1l + k0);
        // dot ~= hi*hi + hi*lo + lo*hi (fp32 accumulate), 2x2 register blocking
        c00 = WMMA_BF16(a0h, b0h, c00);
        c00 = WMMA_BF16(a0h, b0l, c00);
        c00 = WMMA_BF16(a0l, b0h, c00);
        c01 = WMMA_BF16(a0h, b1h, c01);
        c01 = WMMA_BF16(a0h, b1l, c01);
        c01 = WMMA_BF16(a0l, b1h, c01);
        c10 = WMMA_BF16(a1h, b0h, c10);
        c10 = WMMA_BF16(a1h, b0l, c10);
        c10 = WMMA_BF16(a1l, b0h, c10);
        c11 = WMMA_BF16(a1h, b1h, c11);
        c11 = WMMA_BF16(a1h, b1l, c11);
        c11 = WMMA_BF16(a1l, b1h, c11);
    }

    // C/D layout: lane L, VGPR r -> M = r + 8*(L>>4), N = L&15
    int mB = ((lane >> 4) << 3);
    scatter_tile(c00, m0 + mB,      n0 + lr,      sq, targets, posOut, negOut);
    scatter_tile(c01, m0 + mB,      n0 + 16 + lr, sq, targets, posOut, negOut);
    scatter_tile(c10, m0 + 16 + mB, n0 + lr,      sq, targets, posOut, negOut);
    scatter_tile(c11, m0 + 16 + mB, n0 + 16 + lr, sq, targets, posOut, negOut);
}

// ---------------- per-row top-17 selection + loss ----------------

__device__ __forceinline__ void insert17(float (&v)[17], float d) {
    if (d < v[16]) {
        v[16] = d;
#pragma unroll
        for (int j = 16; j > 0; --j) {       // one bubble pass restores sorted order
            float a = v[j - 1], b = v[j];
            v[j - 1] = fminf(a, b);
            v[j]     = fmaxf(a, b);
        }
    }
}

__device__ __forceinline__ float block_sum(float x, float* red, int tid) {
    red[tid] = x; __syncthreads();
    for (int h = 128; h > 0; h >>= 1) {
        if (tid < h) red[tid] += red[tid + h];
        __syncthreads();
    }
    float r = red[0]; __syncthreads();
    return r;
}

__device__ __forceinline__ float block_min(float x, float* red, int tid) {
    red[tid] = x; __syncthreads();
    for (int h = 128; h > 0; h >>= 1) {
        if (tid < h) red[tid] = fminf(red[tid], red[tid + h]);
        __syncthreads();
    }
    float r = red[0]; __syncthreads();
    return r;
}

__global__ void __launch_bounds__(256) select_loss_kernel(
        const float* __restrict__ posD, const float* __restrict__ negD, float* acc) {
    __shared__ float lists[256 * 17];
    __shared__ float red[256];
    int row = blockIdx.x, tid = threadIdx.x;
    const float* pr = posD + (size_t)row * N_POS;
    const float* nr = negD + (size_t)row * N_NEG;

    float v[17];
#pragma unroll
    for (int j = 0; j < 17; ++j) v[j] = 3.0e38f;
    float sp = 0.0f, sn = 0.0f, mp = 3.0e38f;
    for (int idx = tid; idx < N_POS + N_NEG; idx += 256) {
        float d = (idx < N_POS) ? pr[idx] : nr[idx - N_POS];
        if (idx < N_POS) { sp += d; mp = fminf(mp, d); } else { sn += d; }
        insert17(v, d);
    }
#pragma unroll
    for (int j = 0; j < 17; ++j) lists[tid * 17 + j] = v[j];
    __syncthreads();

    if (tid < 32) {                        // each lane merges 8 thread-lists
        float w[17];
#pragma unroll
        for (int j = 0; j < 17; ++j) w[j] = 3.0e38f;
        for (int s = 0; s < 8; ++s)
            for (int j = 0; j < 17; ++j)
                insert17(w, lists[(tid * 8 + s) * 17 + j]);
#pragma unroll
        for (int j = 0; j < 17; ++j) lists[tid * 17 + j] = w[j];
    }
    __syncthreads();
    if (tid == 0) {                        // final merge of 32 lists -> threshold
        float w[17];
#pragma unroll
        for (int j = 0; j < 17; ++j) w[j] = 3.0e38f;
        for (int s = 0; s < 32; ++s)
            for (int j = 0; j < 17; ++j)
                insert17(w, lists[s * 17 + j]);
        red[0] = w[16];                    // 17th smallest == pair_sorted[:,K]
    }
    __syncthreads();
    float thr = red[0];
    __syncthreads();

    float mpAll = block_min(mp, red, tid); // min positive distance (fallback)
    float spAll = block_sum(sp, red, tid); // sum pos distances (pos_d mean)
    float snAll = block_sum(sn, red, tid); // sum neg distances (neg_d mean)

    float ps = 0.0f, ns = 0.0f;
    for (int idx = tid; idx < N_POS + N_NEG; idx += 256) {
        float d = (idx < N_POS) ? pr[idx] : nr[idx - N_POS];
        if (d < thr) {
            float e = expf(ALPHA_C * (1.0f - d));
            if (idx < N_POS) ps += e; else ns += e;
        }
    }
    float psAll = block_sum(ps, red, tid);
    float nsAll = block_sum(ns, red, tid);

    if (tid == 0) {
        float pl = (mpAll < thr) ? psAll : expf(ALPHA_C * (1.0f - mpAll));
        float li = logf((pl + nsAll) / pl);     // -log(pl/(pl+nl))
        atomicAdd(&acc[0], li);
        atomicAdd(&acc[1], (li < 0.6f) ? 1.0f : 0.0f);
        atomicAdd(&acc[2], spAll);
        atomicAdd(&acc[3], snAll);
    }
}

__global__ void finalize_kernel(const float* acc, float* out) {
    if (threadIdx.x == 0) {
        out[0] = acc[0] / (float)N_PTS;
        out[1] = acc[1] / (float)N_PTS;
        out[2] = acc[2] / ((float)N_PTS * (float)N_POS);
        out[3] = acc[3] / ((float)N_PTS * (float)N_NEG);
    }
}

// ---------------- launch ----------------

extern "C" void kernel_launch(void* const* d_in, const int* in_sizes, int n_in,
                              void* d_out, int out_size, void* d_ws, size_t ws_size,
                              hipStream_t stream) {
    const float* x       = (const float*)d_in[0];
    const int*   targets = (const int*)d_in[1];
    float*       out     = (float*)d_out;

    __bf16* xhi = (__bf16*)d_ws;
    __bf16* xlo = xhi + (size_t)N_PTS * D_DIM;
    float*  sq  = (float*)(xlo + (size_t)N_PTS * D_DIM);
    float*  acc = sq + N_PTS;

    float* posOut = out + 4;
    float* negOut = posOut + (size_t)N_PTS * N_POS;

    zero_acc_kernel<<<1, 32, 0, stream>>>(acc);
    cvt_hilo_kernel<<<2048, 256, 0, stream>>>(x, xhi, xlo);
    row_sq_kernel<<<N_PTS, 256, 0, stream>>>(x, sq);
    // 64 (M) x 32 (N) workgroups; each WG = 8 waves covering 64x128 output
    gram_dist_kernel<<<64 * 32, 256, 0, stream>>>(xhi, xlo, sq, targets, posOut, negOut);
    select_loss_kernel<<<N_PTS, 256, 0, stream>>>(posOut, negOut, acc);
    finalize_kernel<<<1, 1, 0, stream>>>(acc, out);
}
